// AttentionWrapper_10668698764010
// MI455X (gfx1250) — matrix operand order repaired and checked by reading, not compiled, not run on hardware
//
#include <hip/hip_runtime.h>
#include <cstdint>
#include <cstddef>

// ---------------- types / helpers ----------------
typedef __attribute__((ext_vector_type(16))) __bf16 v16bf;
typedef __attribute__((ext_vector_type(8)))  float  v8f;

#define NLATc 73
#define NLONc 144
#define Cc    256
#define Hc    4
#define DHc   64
#define Lc    9
#define NPc   (NLATc * NLONc)   // 10512
#define NTILE (NPc / 16)        // 657 exact

union ABu { unsigned int u[8]; v16bf v; };
union C8u { float f[8]; v8f v; };

__device__ __forceinline__ unsigned short f2bf(float f) {
  unsigned int u = __float_as_uint(f);
  unsigned int r = u + 0x7FFFu + ((u >> 16) & 1u);  // round-to-nearest-even
  return (unsigned short)(r >> 16);
}

__device__ __forceinline__ v8f v8zero() {
  v8f z = {0.f, 0.f, 0.f, 0.f, 0.f, 0.f, 0.f, 0.f};
  return z;
}

// ---------------- kernel 0: weights fp32 -> bf16 ----------------
__global__ __launch_bounds__(256) void wconv_kernel(
    const float* __restrict__ wq, const float* __restrict__ wk,
    const float* __restrict__ wv, const float* __restrict__ wo,
    unsigned short* __restrict__ dst) {
  int idx = blockIdx.x * 256 + threadIdx.x;          // 0 .. 262143
  const float* src = (idx < 65536) ? wq
                   : (idx < 131072) ? wk
                   : (idx < 196608) ? wv : wo;
  dst[idx] = f2bf(src[idx & 65535]);
}

// ---------------- kernel 1: LayerNorm + QKV projection ----------------
// grid: 657 blocks of 256 threads (8 waves). 16 points per block.
// outputs q/k/v as bf16 in [H][NP][64] layout.
__global__ __launch_bounds__(256) void ln_qkv_kernel(
    const float* __restrict__ x, const float* __restrict__ ln_w, const float* __restrict__ ln_b,
    const unsigned short* __restrict__ wqb, const unsigned short* __restrict__ wkb,
    const unsigned short* __restrict__ wvb,
    const float* __restrict__ bq, const float* __restrict__ bk, const float* __restrict__ bv,
    unsigned short* __restrict__ q_b, unsigned short* __restrict__ k_b,
    unsigned short* __restrict__ v_b) {
  __shared__ float xt[16][Cc];            // 16 KB
  __shared__ unsigned short xb[16][Cc];   // 8 KB  (bf16 A-matrix source)
  __shared__ float mu_s[16], rs_s[16];

  int tid = threadIdx.x;
  int p0  = blockIdx.x * 16;
  int m   = tid & 15, cb = tid >> 4;

  #pragma unroll 4
  for (int j = 0; j < 16; ++j) {
    int c = cb * 16 + j;
    xt[m][c] = x[(size_t)c * NPc + p0 + m];
  }
  __syncthreads();

  if (tid < 16) {
    float s = 0.f, ss = 0.f;
    for (int c = 0; c < Cc; ++c) { float v = xt[tid][c]; s += v; ss += v * v; }
    float mean = s * (1.0f / Cc);
    float var  = ss * (1.0f / Cc) - mean * mean;
    mu_s[tid] = mean;
    rs_s[tid] = rsqrtf(var + 1e-5f);
  }
  __syncthreads();

  #pragma unroll 4
  for (int j = 0; j < 16; ++j) {
    int c = cb * 16 + j;
    float xn = (xt[m][c] - mu_s[m]) * rs_s[m] * ln_w[c] + ln_b[c];
    xb[m][c] = f2bf(xn);
  }
  __syncthreads();

  int wave = tid >> 5, lane = tid & 31, ln = lane & 15, hi = lane >> 4;

  // 48 output tiles (3 matrices x 16 col-tiles); 6 per wave
  for (int s = 0; s < 6; ++s) {
    int tt  = wave * 6 + s;
    int mat = tt >> 4;
    int o0  = (tt & 15) * 16;
    const unsigned short* wb   = (mat == 0) ? wqb : (mat == 1) ? wkb : wvb;
    const float*          bias = (mat == 0) ? bq  : (mat == 1) ? bk  : bv;
    unsigned short*       dst  = (mat == 0) ? q_b : (mat == 1) ? k_b : v_b;

    v8f acc = v8zero();
    #pragma unroll
    for (int kk = 0; kk < 8; ++kk) {
      ABu A, B;
      const unsigned int* ar = (const unsigned int*)&xb[0][0] + ln * 128 + kk * 16 + hi * 8;
      #pragma unroll
      for (int j = 0; j < 4; ++j) { A.u[j] = ar[j]; A.u[4 + j] = ar[4 + j]; }
      const unsigned int* br = (const unsigned int*)wb + (size_t)(o0 + ln) * 128 + kk * 16 + hi * 8;
      #pragma unroll
      for (int j = 0; j < 8; ++j) B.u[j] = br[j];
      acc = __builtin_amdgcn_wmma_f32_16x16x32_bf16(false, A.v, false, B.v,
                                                    (short)0, acc, false, false);
    }
    int og = o0 + ln;
    int hh = og >> 6, cd = og & 63;
    float bi = bias[og];
    C8u u; u.v = acc;
    #pragma unroll
    for (int r = 0; r < 8; ++r) {
      int p = p0 + r + 8 * hi;
      dst[((size_t)hh * NPc + p) * DHc + cd] = f2bf(u.f[r] + bi);
    }
  }
}

// ---------------- kernel 2: flash-style neighborhood attention ----------------
// grid: H*NLAT = 292 blocks, 288 threads (9 waves; wave w owns queries w*16..w*16+15).
// dynamic LDS layout (bytes):
//   Qs  [144][64] bf16   @ 0      (18432)
//   Ks  [144][64] bf16   @ 18432  (18432)
//   Vt  [64][160] bf16   @ 36864  (20480)   (V transposed, keys padded to 160)
//   Pb  [9][16][160] bf16 @ 57344 (46080)   per-wave softmax tile (A-layout source)
//   total 103424 bytes
__global__ __launch_bounds__(288) void attn_kernel(
    const unsigned short* __restrict__ q_b, const unsigned short* __restrict__ k_b,
    const unsigned short* __restrict__ v_b, const int* __restrict__ rows_idx,
    const float* __restrict__ wquad, const unsigned char* __restrict__ mask,
    unsigned short* __restrict__ oa) {
  extern __shared__ char smem[];
  unsigned short* Qs = (unsigned short*)(smem);
  unsigned short* Ks = (unsigned short*)(smem + 18432);
  unsigned short* Vt = (unsigned short*)(smem + 36864);
  unsigned short* Pb = (unsigned short*)(smem + 57344);

  int blk = blockIdx.x;
  int h = blk / NLATc;
  int i = blk % NLATc;
  int tid = threadIdx.x;
  int wave = tid >> 5, lane = tid & 31, ln = lane & 15, hi = lane >> 4;
  int q0 = wave * 16;
  unsigned short* Pw = Pb + wave * (16 * 160);

  { // load Q tile for this (h, i): 144 rows x 32 dwords
    unsigned int* qd = (unsigned int*)Qs;
    const unsigned int* qs = (const unsigned int*)q_b + ((size_t)h * NPc + (size_t)i * NLONc) * 32;
    for (int t = tid; t < NLONc * 32; t += 288) qd[t] = qs[t];
  }

  float m_run[8], s_run[8], alpha8[8];
  v8f accv[4];
  #pragma unroll
  for (int r = 0; r < 8; ++r) { m_run[r] = -3.0e38f; s_run[r] = 0.f; }
  #pragma unroll
  for (int nt = 0; nt < 4; ++nt) accv[nt] = v8zero();

  const float scale = 0.125f;  // dh^-0.5, dh=64

  for (int l = 0; l < Lc; ++l) {
    __syncthreads();  // previous iteration's Ks/Vt reads done
    int row = rows_idx[i * Lc + l];
    float lw = __logf(wquad[row]);

    { // K row-major, V transposed (+ zero pad cols 144..159)
      unsigned int* kd = (unsigned int*)Ks;
      const unsigned int* ks = (const unsigned int*)k_b + ((size_t)h * NPc + (size_t)row * NLONc) * 32;
      for (int t = tid; t < NLONc * 32; t += 288) kd[t] = ks[t];
      const unsigned int* vs = (const unsigned int*)v_b + ((size_t)h * NPc + (size_t)row * NLONc) * 32;
      for (int t = tid; t < NLONc * 32; t += 288) {
        int y = t >> 5, cd2 = t & 31;
        unsigned int d = vs[t];
        int c = cd2 * 2;
        Vt[c * 160 + y]       = (unsigned short)(d & 0xFFFFu);
        Vt[(c + 1) * 160 + y] = (unsigned short)(d >> 16);
      }
      for (int t = tid; t < 64 * 16; t += 288) {
        int c = t >> 4, y = 144 + (t & 15);
        Vt[c * 160 + y] = 0;
      }
    }
    __syncthreads();

    // ---- pass 1: S = Q K^T (wave's 16 rows x 144 cols, in registers) ----
    ABu A0, A1;
    {
      const unsigned int* ar = (const unsigned int*)Qs + (q0 + ln) * 32 + hi * 8;
      #pragma unroll
      for (int j = 0; j < 4; ++j) {
        A0.u[j] = ar[j];      A0.u[4 + j] = ar[4 + j];
        A1.u[j] = ar[16 + j]; A1.u[4 + j] = ar[20 + j];
      }
    }
    float sc[9][8];
    #pragma unroll
    for (int t9 = 0; t9 < 9; ++t9) {
      v8f c8 = v8zero();
      ABu B;
      const unsigned int* br = (const unsigned int*)Ks + (t9 * 16 + ln) * 32 + hi * 8;
      #pragma unroll
      for (int j = 0; j < 8; ++j) B.u[j] = br[j];
      c8 = __builtin_amdgcn_wmma_f32_16x16x32_bf16(false, A0.v, false, B.v, (short)0, c8, false, false);
      #pragma unroll
      for (int j = 0; j < 8; ++j) B.u[j] = br[16 + j];
      c8 = __builtin_amdgcn_wmma_f32_16x16x32_bf16(false, A1.v, false, B.v, (short)0, c8, false, false);
      C8u u; u.v = c8;
      #pragma unroll
      for (int r = 0; r < 8; ++r) sc[t9][r] = u.f[r];
    }

    // ---- mask + log-quadrature + online softmax; write P (bf16) ----
    #pragma unroll
    for (int r = 0; r < 8; ++r) {
      int xq = q0 + r + 8 * hi;
      const unsigned char* mrow = mask + (((size_t)(i * NLONc + xq) * Lc + l) * NLONc);
      float vmax = -3.0e38f;
      #pragma unroll
      for (int t9 = 0; t9 < 9; ++t9) {
        int y = t9 * 16 + ln;
        float val = sc[t9][r] * scale + lw;
        if (!mrow[y]) val = -1.0e30f;
        sc[t9][r] = val;
        vmax = fmaxf(vmax, val);
      }
      vmax = fmaxf(vmax, __shfl_xor(vmax, 1));
      vmax = fmaxf(vmax, __shfl_xor(vmax, 2));
      vmax = fmaxf(vmax, __shfl_xor(vmax, 4));
      vmax = fmaxf(vmax, __shfl_xor(vmax, 8));
      float nm    = fmaxf(m_run[r], vmax);
      float alpha = __expf(m_run[r] - nm);
      float ps = 0.f;
      #pragma unroll
      for (int t9 = 0; t9 < 9; ++t9) {
        float p = __expf(sc[t9][r] - nm);
        ps += p;
        Pw[(r + 8 * hi) * 160 + t9 * 16 + ln] = f2bf(p);
      }
      ps += __shfl_xor(ps, 1);
      ps += __shfl_xor(ps, 2);
      ps += __shfl_xor(ps, 4);
      ps += __shfl_xor(ps, 8);
      s_run[r]  = s_run[r] * alpha + ps;
      m_run[r]  = nm;
      alpha8[r] = alpha;
    }
    #pragma unroll
    for (int r = 0; r < 8; ++r) Pw[(r + 8 * hi) * 160 + 144 + ln] = 0;  // key pad

    #pragma unroll
    for (int nt = 0; nt < 4; ++nt) {  // rescale accumulator
      C8u u; u.v = accv[nt];
      #pragma unroll
      for (int r = 0; r < 8; ++r) u.f[r] *= alpha8[r];
      accv[nt] = u.v;
    }

    // ---- pass 2: acc += P @ V  (contraction over 160 padded keys) ----
    #pragma unroll
    for (int kc = 0; kc < 5; ++kc) {
      ABu A;
      const unsigned int* pr = (const unsigned int*)Pw + ln * 80 + kc * 16 + hi * 8;
      #pragma unroll
      for (int j = 0; j < 4; ++j) { A.u[j] = pr[j]; A.u[4 + j] = pr[4 + j]; }
      #pragma unroll
      for (int nt = 0; nt < 4; ++nt) {
        ABu B;
        const unsigned int* br = (const unsigned int*)Vt + (nt * 16 + ln) * 80 + kc * 16 + hi * 8;
        #pragma unroll
        for (int j = 0; j < 8; ++j) B.u[j] = br[j];
        accv[nt] = __builtin_amdgcn_wmma_f32_16x16x32_bf16(false, A.v, false, B.v,
                                                           (short)0, accv[nt], false, false);
      }
    }
  }

  // ---- finalize: out = acc / rowsum, store bf16 [N][256] (head-concat) ----
  #pragma unroll
  for (int nt = 0; nt < 4; ++nt) {
    C8u u; u.v = accv[nt];
    #pragma unroll
    for (int r = 0; r < 8; ++r) {
      int xq = q0 + r + 8 * hi;
      float o = u.f[r] / s_run[r];
      oa[(size_t)(i * NLONc + xq) * Cc + h * DHc + nt * 16 + ln] = f2bf(o);
    }
  }
}

// ---------------- kernel 3: output projection + bias + residual ----------------
__global__ __launch_bounds__(256) void oproj_kernel(
    const unsigned short* __restrict__ oa, const unsigned short* __restrict__ wob,
    const float* __restrict__ bo, const float* __restrict__ x, float* __restrict__ out) {
  __shared__ unsigned short ab[16][Cc];  // 8 KB
  int tid = threadIdx.x;
  int p0  = blockIdx.x * 16;

  unsigned int* abd = (unsigned int*)&ab[0][0];
  const unsigned int* src = (const unsigned int*)oa + (size_t)p0 * 128;
  for (int t = tid; t < 16 * 128; t += 256) abd[t] = src[t];
  __syncthreads();

  int wave = tid >> 5, lane = tid & 31, ln = lane & 15, hi = lane >> 4;
  for (int s = 0; s < 2; ++s) {
    int o0 = (wave * 2 + s) * 16;
    v8f acc = v8zero();
    #pragma unroll
    for (int kk = 0; kk < 8; ++kk) {
      ABu A, B;
      const unsigned int* ar = (const unsigned int*)&ab[0][0] + ln * 128 + kk * 16 + hi * 8;
      #pragma unroll
      for (int j = 0; j < 4; ++j) { A.u[j] = ar[j]; A.u[4 + j] = ar[4 + j]; }
      const unsigned int* br = (const unsigned int*)wob + (size_t)(o0 + ln) * 128 + kk * 16 + hi * 8;
      #pragma unroll
      for (int j = 0; j < 8; ++j) B.u[j] = br[j];
      acc = __builtin_amdgcn_wmma_f32_16x16x32_bf16(false, A.v, false, B.v,
                                                    (short)0, acc, false, false);
    }
    int og = o0 + ln;
    float bi = bo[og];
    C8u u; u.v = acc;
    #pragma unroll
    for (int r = 0; r < 8; ++r) {
      int p = p0 + r + 8 * hi;
      out[(size_t)og * NPc + p] = x[(size_t)og * NPc + p] + bi + u.f[r];
    }
  }
}

// ---------------- launch ----------------
extern "C" void kernel_launch(void* const* d_in, const int* in_sizes, int n_in,
                              void* d_out, int out_size, void* d_ws, size_t ws_size,
                              hipStream_t stream) {
  const float* x     = (const float*)d_in[0];
  const float* ln_w  = (const float*)d_in[1];
  const float* ln_b  = (const float*)d_in[2];
  const float* wq    = (const float*)d_in[3];
  const float* bq    = (const float*)d_in[4];
  const float* wk    = (const float*)d_in[5];
  const float* bk    = (const float*)d_in[6];
  const float* wv    = (const float*)d_in[7];
  const float* bv    = (const float*)d_in[8];
  const float* wo    = (const float*)d_in[9];
  const float* bo    = (const float*)d_in[10];
  const float* wquad = (const float*)d_in[11];
  const int* rows_idx = (const int*)d_in[12];
  const unsigned char* mask = (const unsigned char*)d_in[13];
  float* out = (float*)d_out;

  // workspace layout (bytes)
  char* ws = (char*)d_ws;
  const size_t WSZ = 65536ull * 2;            // one bf16 weight matrix
  const size_t QSZ = (size_t)Hc * NPc * DHc * 2;  // 5,382,144
  unsigned short* wqb = (unsigned short*)(ws);
  unsigned short* wkb = (unsigned short*)(ws + WSZ);
  unsigned short* wvb = (unsigned short*)(ws + 2 * WSZ);
  unsigned short* wob = (unsigned short*)(ws + 3 * WSZ);
  unsigned short* q_b = (unsigned short*)(ws + 4 * WSZ);
  unsigned short* k_b = (unsigned short*)(ws + 4 * WSZ + QSZ);
  unsigned short* v_b = (unsigned short*)(ws + 4 * WSZ + 2 * QSZ);
  unsigned short* oab = (unsigned short*)(ws + 4 * WSZ + 3 * QSZ);

  wconv_kernel<<<1024, 256, 0, stream>>>(wq, wk, wv, wo, wqb);
  ln_qkv_kernel<<<NTILE, 256, 0, stream>>>(x, ln_w, ln_b, wqb, wkb, wvb,
                                           bq, bk, bv, q_b, k_b, v_b);
  attn_kernel<<<Hc * NLATc, 288, 103424, stream>>>(q_b, k_b, v_b, rows_idx,
                                                   wquad, mask, oab);
  oproj_kernel<<<NTILE, 256, 0, stream>>>(oab, wob, bo, x, out);
}
